// CodanoBlockND_26568667693786
// MI455X (gfx1250) — matrix operand order repaired and checked by reading, not compiled, not run on hardware
//
#include <hip/hip_runtime.h>
#include <hip/hip_bf16.h>

// ---------------- problem constants (from reference setup_inputs) -----------
constexpr int H_  = 128;
constexpr int W_  = 128;
constexpr int HW_ = H_ * W_;          // 16384
constexpr int KC  = 56;               // channels / tokens
constexpr int B_  = 2;
constexpr int A_  = 8;                // heads
constexpr int BK  = B_ * KC;          // 112 images
constexpr int RM  = 32;               // retained rows: 0..15 and 112..127
constexpr int CM  = 16;               // retained rfft cols: 0..15
constexpr float EPSI = 1e-5f;

typedef __attribute__((ext_vector_type(16))) __bf16 v16bf;
typedef __attribute__((ext_vector_type(8)))  __bf16 v8bf;
typedef __attribute__((ext_vector_type(8)))  float  v8f;

__device__ __forceinline__ v16bf cat16(v8bf lo, v8bf hi) {
    return __builtin_shufflevector(lo, hi, 0,1,2,3,4,5,6,7,8,9,10,11,12,13,14,15);
}

// ---------------- instance norm (block per 128x128 plane) -------------------
// out = inorm(a + badd?) * g + bb + res?
__global__ __launch_bounds__(256)
void inorm_k(const float* __restrict__ a, const float* __restrict__ badd,
             const float* __restrict__ g, const float* __restrict__ bb,
             const float* __restrict__ res, float* __restrict__ out) {
    const int plane = blockIdx.x;
    const int tid = threadIdx.x;
    const float* pa = a + (size_t)plane * HW_;
    const float* pb = badd ? badd + (size_t)plane * HW_ : nullptr;
    const float* pr = res  ? res  + (size_t)plane * HW_ : nullptr;
    float* po = out + (size_t)plane * HW_;

    __shared__ float s1[256], s2[256];
    float sum = 0.f, ssq = 0.f;
    for (int i = tid; i < HW_; i += 256) {
        float v = pa[i] + (pb ? pb[i] : 0.f);
        sum += v; ssq += v * v;
    }
    s1[tid] = sum; s2[tid] = ssq;
    __syncthreads();
    for (int s = 128; s > 0; s >>= 1) {
        if (tid < s) { s1[tid] += s1[tid + s]; s2[tid] += s2[tid + s]; }
        __syncthreads();
    }
    const float mu  = s1[0] * (1.f / HW_);
    const float var = s2[0] * (1.f / HW_) - mu * mu;
    const float rs  = rsqrtf(var + EPSI);
    const float gg  = g  ? g[0]  : 1.f;
    const float be  = bb ? bb[0] : 0.f;
    for (int i = tid; i < HW_; i += 256) {
        float v = pa[i] + (pb ? pb[i] : 0.f);
        v = (v - mu) * rs * gg + be;
        if (pr) v += pr[i];
        po[i] = v;
    }
}

// ---------------- forward truncated DFT: plane -> 32x16 complex modes -------
// Xf[(plane*512 + r*16 + c)*2 + {re,im}], rows r<16 -> freq r, r>=16 -> freq 96+r
__global__ __launch_bounds__(256)
void fwd_dft_k(const float* __restrict__ in, float* __restrict__ Xf) {
    const int plane = blockIdx.x;
    const int tid = threadIdx.x;
    const float* base = in + (size_t)plane * HW_;

    __shared__ float ct[128], st[128];
    __shared__ float Tre[H_ * CM], Tim[H_ * CM];   // 128x16 complex, 16 KB
    if (tid < 128) {
        float ang = 6.28318530717958647692f * (float)tid / 128.f;
        ct[tid] = __cosf(ang); st[tid] = __sinf(ang);
    }
    __syncthreads();

    // stage A: T[h,c] = sum_w x[h,w] * exp(-i*2pi*c*w/128)
    for (int idx = tid; idx < H_ * CM; idx += 256) {
        const int h = idx >> 4, c = idx & 15;
        const float* row = base + h * W_;
        float re = 0.f, im = 0.f;
        for (int w = 0; w < W_; ++w) {
            const float xv = row[w];
            const int k = (c * w) & 127;
            re += xv * ct[k];
            im -= xv * st[k];
        }
        Tre[idx] = re; Tim[idx] = im;
    }
    __syncthreads();

    // stage B: Xf[r,c] = (1/HW) * sum_h T[h,c] * exp(-i*2pi*f(r)*h/128)
    for (int idx = tid; idx < RM * CM; idx += 256) {
        const int r = idx >> 4, c = idx & 15;
        const int f = (r < 16) ? r : (96 + r);     // 112..127
        float re = 0.f, im = 0.f;
        for (int h = 0; h < H_; ++h) {
            const float tr = Tre[h * CM + c], ti = Tim[h * CM + c];
            const int k = (f * h) & 127;
            re += tr * ct[k] + ti * st[k];
            im += ti * ct[k] - tr * st[k];
        }
        float* o = Xf + ((size_t)plane * (RM * CM) + idx) * 2;
        o[0] = re * (1.f / HW_);
        o[1] = im * (1.f / HW_);
    }
}

// ---------------- per-mode channel mixing: Yf = W (x) Xf --------------------
__global__ __launch_bounds__(256)
void mode_mix_k(const float* __restrict__ Xf, const float* __restrict__ wspec,
                float* __restrict__ Yf, int nimg, int Ci, int Co) {
    const int total = nimg * Co * (RM * CM);
    const int t = blockIdx.x * 256 + threadIdx.x;
    if (t >= total) return;
    const int rc  = t & 511;
    const int tmp = t >> 9;
    const int o   = tmp % Co;
    const int img = tmp / Co;
    const int r = rc >> 4, c = rc & 15;
    const int br = (r < 16) ? 0 : 1;
    const int rp = (r < 16) ? r : (r - 16);
    float yr = 0.f, yi = 0.f;
    for (int i = 0; i < Ci; ++i) {
        const float* xp = Xf + ((size_t)(img * Ci + i) * (RM * CM) + rc) * 2;
        const float* wp = wspec + ((size_t)(((br * Ci + i) * Co + o) * 16 + rp) * 16 + c) * 2;
        const float ar = xp[0], ai = xp[1], wr = wp[0], wi = wp[1];
        yr += ar * wr - ai * wi;
        yi += ar * wi + ai * wr;
    }
    float* yp = Yf + ((size_t)(img * Co + o) * (RM * CM) + rc) * 2;
    yp[0] = yr; yp[1] = yi;
}

// ---------------- inverse truncated DFT (+bias, +1x1 skip) ------------------
// block = img*Co + o.  out_bf16: write into padded (B, A, 64, HW) bf16 layout.
__global__ __launch_bounds__(256)
void inv_dft_k(const float* __restrict__ Yf, const float* __restrict__ bias,
               const float* __restrict__ skin, const float* __restrict__ sw,
               const float* __restrict__ sb, void* __restrict__ out,
               int Ci, int Co, int use_skip, int out_bf16) {
    const int blk = blockIdx.x;
    const int img = blk / Co;
    const int o   = blk % Co;
    const int tid = threadIdx.x;
    const float* Y = Yf + (size_t)blk * (RM * CM * 2);

    __shared__ float ct[128], st[128];
    __shared__ float Ure[RM * W_], Uim[RM * W_];   // 32x128 complex, 32 KB
    if (tid < 128) {
        float ang = 6.28318530717958647692f * (float)tid / 128.f;
        ct[tid] = __cosf(ang); st[tid] = __sinf(ang);
    }
    __syncthreads();

    // stage U: U[r,w] = sum_c m(c) * Yf[r,c] * exp(+i*2pi*c*w/128)
    for (int idx = tid; idx < RM * W_; idx += 256) {
        const int r = idx >> 7, w = idx & 127;
        float re = 0.f, im = 0.f;
        for (int c = 0; c < CM; ++c) {
            const float yr = Y[(r * CM + c) * 2], yi = Y[(r * CM + c) * 2 + 1];
            const float m = (c == 0) ? 1.f : 2.f;  // Hermitian completion weight
            const int k = (c * w) & 127;
            re += m * (yr * ct[k] - yi * st[k]);
            im += m * (yr * st[k] + yi * ct[k]);
        }
        Ure[idx] = re; Uim[idx] = im;
    }
    __syncthreads();

    const float bo = bias ? bias[o] : 0.f;
    for (int idx = tid; idx < HW_; idx += 256) {
        const int h = idx >> 7, w = idx & 127;
        float acc = 0.f;
        for (int r = 0; r < RM; ++r) {
            const int f = (r < 16) ? r : (96 + r);
            const int k = (f * h) & 127;
            acc += Ure[r * W_ + w] * ct[k] - Uim[r * W_ + w] * st[k];
        }
        acc += bo;
        if (use_skip) {
            for (int i = 0; i < Ci; ++i)
                acc += skin[((size_t)img * Ci + i) * HW_ + idx] * sw[o * Ci + i];
            acc += sb[o];
        }
        if (out_bf16) {
            const int b = img / KC, kch = img % KC;
            ((__bf16*)out)[(((size_t)(b * A_ + o)) * 64 + kch) * HW_ + idx] = (__bf16)acc;
        } else {
            ((float*)out)[((size_t)img * Co + o) * HW_ + idx] = acc;
        }
    }
}

// ---------------- attention scores: Q @ K^T over HW (bf16 WMMA) -------------
// one block per (b,a); 8 waves; wave w -> output tiles 2w, 2w+1 of a 64x64 grid
__global__ __launch_bounds__(256)
void scores_k(const __bf16* __restrict__ qb, const __bf16* __restrict__ kb,
              float* __restrict__ sc) {
    const int ba   = blockIdx.x;
    const int lane = threadIdx.x & 31;
    const int wv   = threadIdx.x >> 5;
    const int t0   = 2 * wv;
    const int mt   = t0 >> 2;
    const int nt0  = t0 & 3, nt1 = nt0 + 1;
    const int lm   = lane & 15, sel = lane >> 4;

    const __bf16* qbase = qb + (size_t)ba * 64 * HW_;
    const __bf16* kbase = kb + (size_t)ba * 64 * HW_;
    const __bf16* arow  = qbase + (size_t)(mt  * 16 + lm) * HW_;
    const __bf16* brow0 = kbase + (size_t)(nt0 * 16 + lm) * HW_;
    const __bf16* brow1 = kbase + (size_t)(nt1 * 16 + lm) * HW_;

    v8f c0 = {0.f,0.f,0.f,0.f,0.f,0.f,0.f,0.f};
    v8f c1 = {0.f,0.f,0.f,0.f,0.f,0.f,0.f,0.f};

    for (int kk = 0; kk < HW_; kk += 32) {
        if ((kk & 255) == 0 && kk + 512 < HW_) {        // global_prefetch_b8
            __builtin_prefetch(arow  + kk + 512, 0, 3);
            __builtin_prefetch(brow0 + kk + 512, 0, 3);
            __builtin_prefetch(brow1 + kk + 512, 0, 3);
        }
        v8bf alo = *(const v8bf*)(arow + kk + sel * 16);
        v8bf ahi = *(const v8bf*)(arow + kk + sel * 16 + 8);
        v16bf Am = cat16(alo, ahi);
        v16bf B0 = *(const v16bf*)(brow0 + kk + sel * 16);
        v16bf B1 = *(const v16bf*)(brow1 + kk + sel * 16);
        c0 = __builtin_amdgcn_wmma_f32_16x16x32_bf16(false, Am, false, B0, (short)0, c0, false, false);
        c1 = __builtin_amdgcn_wmma_f32_16x16x32_bf16(false, Am, false, B1, (short)0, c1, false, false);
    }
    float* out = sc + (size_t)ba * 4096;
    for (int v = 0; v < 8; ++v) {
        const int m = mt * 16 + v + sel * 8;
        out[m * 64 + nt0 * 16 + lm] = c0[v];
        out[m * 64 + nt1 * 16 + lm] = c1[v];
    }
}

// ---------------- softmax over 56 valid columns, emit padded bf16 -----------
__global__ __launch_bounds__(64)
void softmax_k(const float* __restrict__ sc, __bf16* __restrict__ attnb) {
    const int ba = blockIdx.x;
    const int r  = threadIdx.x;             // 0..63
    const float* row = sc + (size_t)ba * 4096 + r * 64;
    __bf16* orow = attnb + (size_t)ba * 4096 + r * 64;
    if (r >= KC) { for (int j = 0; j < 64; ++j) orow[j] = (__bf16)0.f; return; }
    const float scale = 1.f / 128.f;        // 1/sqrt(HW)
    float mx = -3.4e38f;
    for (int j = 0; j < KC; ++j) mx = fmaxf(mx, row[j] * scale);
    float s = 0.f;
    for (int j = 0; j < KC; ++j) s += __expf(row[j] * scale - mx);
    const float inv = 1.f / s;
    for (int j = 0; j < KC; ++j) orow[j] = (__bf16)(__expf(row[j] * scale - mx) * inv);
    for (int j = KC; j < 64; ++j) orow[j] = (__bf16)0.f;
}

// ---------------- out = attn @ V  (bf16 WMMA), scatter to (BK, A, HW) -------
__global__ __launch_bounds__(256)
void av_k(const __bf16* __restrict__ attnb, const __bf16* __restrict__ vb,
          float* __restrict__ avout) {
    const int ba   = blockIdx.y;
    const int lane = threadIdx.x & 31;
    const int wv   = threadIdx.x >> 5;
    const int tile = blockIdx.x * 8 + wv;   // 0..4095
    const int mt   = tile & 3;
    const int stt  = tile >> 2;             // spatial tile 0..1023
    const int lm   = lane & 15, sel = lane >> 4;

    const __bf16* abase = attnb + (size_t)ba * 4096 + (mt * 16 + lm) * 64;
    const __bf16* vbase = vb + (size_t)ba * 64 * HW_;
    const int s = stt * 16 + lm;

    v8f c = {0.f,0.f,0.f,0.f,0.f,0.f,0.f,0.f};
    for (int kk = 0; kk < 64; kk += 32) {
        v8bf alo = *(const v8bf*)(abase + kk + sel * 16);
        v8bf ahi = *(const v8bf*)(abase + kk + sel * 16 + 8);
        v16bf Am = cat16(alo, ahi);
        v16bf Bv = {};
        const int l0 = kk + sel * 16;
        for (int j = 0; j < 16; ++j) Bv[j] = vbase[(size_t)(l0 + j) * HW_ + s];
        c = __builtin_amdgcn_wmma_f32_16x16x32_bf16(false, Am, false, Bv, (short)0, c, false, false);
    }
    const int b = ba >> 3, a = ba & 7;
    for (int v = 0; v < 8; ++v) {
        const int m = mt * 16 + v + sel * 8;
        if (m < KC)
            avout[(((size_t)(b * KC + m)) * A_ + a) * HW_ + s] = c[v];
    }
}

// ---------------- elementwise skip + optional exact GELU --------------------
__global__ __launch_bounds__(256)
void skipact_k(const float* __restrict__ a, const float* __restrict__ xin,
               const float* __restrict__ sw, const float* __restrict__ sb,
               float* __restrict__ out, int n, int act) {
    const int i = blockIdx.x * 256 + threadIdx.x;
    if (i >= n) return;
    float v = a[i] + xin[i] * sw[0] + sb[0];
    if (act) v = 0.5f * v * (1.f + erff(v * 0.70710678118654752f));
    out[i] = v;
}

// ============================================================================
extern "C" void kernel_launch(void* const* d_in, const int* in_sizes, int n_in,
                              void* d_out, int out_size, void* d_ws, size_t ws_size,
                              hipStream_t stream) {
    (void)in_sizes; (void)n_in; (void)out_size; (void)ws_size;
    // setup_inputs() dict order
    const float* x    = (const float*)d_in[0];
    const float* Kw   = (const float*)d_in[1];
    const float* Kb_  = (const float*)d_in[2];
    const float* Ksw  = (const float*)d_in[3];
    const float* Ksb  = (const float*)d_in[4];
    const float* Qw   = (const float*)d_in[5];
    const float* Qb   = (const float*)d_in[6];
    const float* Qsw  = (const float*)d_in[7];
    const float* Qsb  = (const float*)d_in[8];
    const float* Vw   = (const float*)d_in[9];
    const float* Vb   = (const float*)d_in[10];
    const float* Vsw  = (const float*)d_in[11];
    const float* Vsb  = (const float*)d_in[12];
    const float* Pw   = (const float*)d_in[13];
    const float* Pb   = (const float*)d_in[14];
    const float* Psw  = (const float*)d_in[15];
    const float* Psb  = (const float*)d_in[16];
    const float* M0w  = (const float*)d_in[17];
    const float* M0b  = (const float*)d_in[18];
    const float* M0sw = (const float*)d_in[19];
    const float* M0sb = (const float*)d_in[20];
    const float* M1w  = (const float*)d_in[21];
    const float* M1b  = (const float*)d_in[22];
    const float* M1sw = (const float*)d_in[23];
    const float* M1sb = (const float*)d_in[24];
    const float* g1   = (const float*)d_in[25];
    const float* g2   = (const float*)d_in[26];
    const float* gan  = (const float*)d_in[27];
    const float* gmo  = (const float*)d_in[28];
    const float* b1   = (const float*)d_in[29];
    const float* b2   = (const float*)d_in[30];
    const float* ban  = (const float*)d_in[31];
    const float* bmo  = (const float*)d_in[32];

    // workspace bump allocator (256B aligned)
    char* wsp = (char*)d_ws;
    size_t off = 0;
    auto alloc = [&](size_t bytes) -> void* {
        void* p = wsp + off;
        off = (off + bytes + 255) & ~size_t(255);
        return p;
    };
    const size_t planeB = (size_t)HW_ * sizeof(float);
    float*  xn    = (float*)alloc(BK * planeB);
    float*  Xf    = (float*)alloc((size_t)BK * A_ * RM * CM * 2 * sizeof(float));
    float*  Yf    = (float*)alloc((size_t)BK * A_ * RM * CM * 2 * sizeof(float));
    const size_t padQKV = (size_t)B_ * A_ * 64 * HW_ * sizeof(__bf16);
    __bf16* kbuf  = (__bf16*)alloc(padQKV);
    __bf16* qbuf  = (__bf16*)alloc(padQKV);
    __bf16* vbuf  = (__bf16*)alloc(padQKV);
    float*  sc    = (float*)alloc((size_t)B_ * A_ * 64 * 64 * sizeof(float));
    __bf16* attnb = (__bf16*)alloc((size_t)B_ * A_ * 64 * 64 * sizeof(__bf16));
    float*  avout = (float*)alloc((size_t)BK * A_ * HW_ * sizeof(float));
    float*  yproj = (float*)alloc(BK * planeB);
    float*  att   = (float*)alloc(BK * planeB);
    float*  an    = (float*)alloc(BK * planeB);
    float*  t1    = (float*)alloc(BK * planeB);
    float*  t2    = (float*)alloc(BK * planeB);
    float*  hb    = (float*)alloc(BK * planeB);
    float*  hb2   = (float*)alloc(BK * planeB);

    // zero padded bf16 Q/K/V (rows 56..63 stay zero; valid rows overwritten)
    hipMemsetAsync(kbuf, 0, padQKV, stream);
    hipMemsetAsync(qbuf, 0, padQKV, stream);
    hipMemsetAsync(vbuf, 0, padQKV, stream);

    const int elemN = BK * HW_;
    const int mixThreads = 256;

    // 1) xn = inorm(x; g1,b1)
    inorm_k<<<BK, 256, 0, stream>>>(x, nullptr, g1, b1, nullptr, xn);

    // 2) shared forward modes of xn
    fwd_dft_k<<<BK, 256, 0, stream>>>(xn, Xf);

    // 3) K/Q/V FNO layers (Ci=1 -> Co=8), write bf16 padded head layouts
    {
        int total = BK * A_ * RM * CM;
        int mblocks = (total + mixThreads - 1) / mixThreads;
        // K -> kbuf
        mode_mix_k<<<mblocks, 256, 0, stream>>>(Xf, Kw, Yf, BK, 1, A_);
        inv_dft_k<<<BK * A_, 256, 0, stream>>>(Yf, Kb_, xn, Ksw, Ksb, kbuf, 1, A_, 1, 1);
        // Q -> qbuf
        mode_mix_k<<<mblocks, 256, 0, stream>>>(Xf, Qw, Yf, BK, 1, A_);
        inv_dft_k<<<BK * A_, 256, 0, stream>>>(Yf, Qb, xn, Qsw, Qsb, qbuf, 1, A_, 1, 1);
        // V -> vbuf
        mode_mix_k<<<mblocks, 256, 0, stream>>>(Xf, Vw, Yf, BK, 1, A_);
        inv_dft_k<<<BK * A_, 256, 0, stream>>>(Yf, Vb, xn, Vsw, Vsb, vbuf, 1, A_, 1, 1);
    }

    // 4) scores = Q K^T  (WMMA bf16)
    scores_k<<<B_ * A_, 256, 0, stream>>>(qbuf, kbuf, sc);

    // 5) softmax -> padded bf16 attention
    softmax_k<<<B_ * A_, 64, 0, stream>>>(sc, attnb);

    // 6) out = attn @ V  (WMMA bf16) -> (BK, A, HW) f32
    av_k<<<dim3(512, B_ * A_), 256, 0, stream>>>(attnb, vbuf, avout);

    // 7) projection FNO layer (Ci=8 -> Co=1)
    fwd_dft_k<<<BK * A_, 256, 0, stream>>>(avout, Xf);
    {
        int total = BK * 1 * RM * CM;
        int mblocks = (total + mixThreads - 1) / mixThreads;
        mode_mix_k<<<mblocks, 256, 0, stream>>>(Xf, Pw, Yf, BK, A_, 1);
        inv_dft_k<<<BK, 256, 0, stream>>>(Yf, Pb, avout, Psw, Psb, yproj, A_, 1, 1, 0);
    }

    // 8) att = inorm(yproj + x; gan,ban) ; 9) an = inorm(att; g2,b2)
    inorm_k<<<BK, 256, 0, stream>>>(yproj, x, gan, ban, nullptr, att);
    inorm_k<<<BK, 256, 0, stream>>>(att, nullptr, g2, b2, nullptr, an);

    // 10) mixer layer 0: inorm(spectral(an)) + skip(an), gelu
    fwd_dft_k<<<BK, 256, 0, stream>>>(an, Xf);
    {
        int total = BK * RM * CM;
        int mblocks = (total + mixThreads - 1) / mixThreads;
        mode_mix_k<<<mblocks, 256, 0, stream>>>(Xf, M0w, Yf, BK, 1, 1);
        inv_dft_k<<<BK, 256, 0, stream>>>(Yf, M0b, nullptr, nullptr, nullptr, t1, 1, 1, 0, 0);
        inorm_k<<<BK, 256, 0, stream>>>(t1, nullptr, nullptr, nullptr, nullptr, t2);
        skipact_k<<<(elemN + 255) / 256, 256, 0, stream>>>(t2, an, M0sw, M0sb, hb, elemN, 1);
    }

    // 11) mixer layer 1: inorm(spectral(hb)) + skip(hb)
    fwd_dft_k<<<BK, 256, 0, stream>>>(hb, Xf);
    {
        int total = BK * RM * CM;
        int mblocks = (total + mixThreads - 1) / mixThreads;
        mode_mix_k<<<mblocks, 256, 0, stream>>>(Xf, M1w, Yf, BK, 1, 1);
        inv_dft_k<<<BK, 256, 0, stream>>>(Yf, M1b, nullptr, nullptr, nullptr, t1, 1, 1, 0, 0);
        inorm_k<<<BK, 256, 0, stream>>>(t1, nullptr, nullptr, nullptr, nullptr, t2);
        skipact_k<<<(elemN + 255) / 256, 256, 0, stream>>>(t2, hb, M1sw, M1sb, hb2, elemN, 0);
    }

    // 12) y = inorm(hb2; gmo,bmo) + att  -> d_out
    inorm_k<<<BK, 256, 0, stream>>>(hb2, nullptr, gmo, bmo, att, (float*)d_out);
}
